// SimpleDecoder_6854767804731
// MI455X (gfx1250) — matrix operand order repaired and checked by reading, not compiled, not run on hardware
//
#include <hip/hip_runtime.h>
#include <hip/hip_bf16.h>

typedef __attribute__((ext_vector_type(16))) _Float16 v16h;
typedef __attribute__((ext_vector_type(8)))  _Float16 v8h;
typedef __attribute__((ext_vector_type(8)))  float    v8f;

#define Bc   32
#define Tc   100
#define Sc   512
#define Hc   512
#define ENCc 1024
#define Vc   8192
#define NHc  4
#define HDc  128

// ---------------------------------------------------------------------------
// WMMA fragment helpers (CDNA5 16-bit layouts, CDNA5 ISA 7.12.2)
// A (16x32, MxK): lanes 0-15 row M=lane, elems 0-7 = K kb..kb+7, 8-15 = kb+16..23
//                 lanes 16-31 row M=lane-16, elems 0-7 = kb+8..15, 8-15 = kb+24..31
// B (32x16, KxN): lanes 0-15 col N=lane, elems 0-7 = K kb..kb+7, 8-15 = kb+16..23
//                 lanes 16-31 col N=lane-16, elems 0-7 = kb+8..15, 8-15 = kb+24..31
// C/D (16x16 f32): lane<16 -> VGPR r = (M=r, N=lane); lane>=16 -> (M=8+r, N=lane-16)
// ---------------------------------------------------------------------------
__device__ __forceinline__ v16h load_frag_a(const _Float16* __restrict__ tile,
                                            long lda, int lane, int kb) {
  int r   = lane & 15;
  int klo = kb + ((lane & 16) ? 8 : 0);
  const _Float16* p = tile + (long)r * lda + klo;
  v8h lo = *(const v8h*)(p);
  v8h hi = *(const v8h*)(p + 16);
  v16h a;
#pragma unroll
  for (int i = 0; i < 8; ++i) { a[i] = lo[i]; a[i + 8] = hi[i]; }
  return a;
}

// W stored row-major [N, K]; row n of W supplies output column n (B = W^T)
__device__ __forceinline__ v16h load_frag_b(const _Float16* __restrict__ wt,
                                            long ldw, int lane, int kb) {
  int n  = lane & 15;
  int kg = kb + ((lane & 16) ? 8 : 0);
  const _Float16* p = wt + (long)n * ldw + kg;
  v8h lo = *(const v8h*)(p);
  v8h hi = *(const v8h*)(p + 16);
  v16h b;
#pragma unroll
  for (int i = 0; i < 8; ++i) { b[i] = lo[i]; b[i + 8] = hi[i]; }
  return b;
}

__device__ __forceinline__ v8f wmma16(v16h a, v16h b, v8f c) {
  return __builtin_amdgcn_wmma_f32_16x16x32_f16(false, a, false, b,
                                                (short)0, c, false, false);
}

__device__ __forceinline__ float sigf(float x) { return 1.0f / (1.0f + __expf(-x)); }

// ---------------------------------------------------------------------------
// Register-blocked GEMM: out = act(A[M,K] * W[N,K]^T + bias).
// Wave computes 32x64 of C via 2x4 accumulator tiles (8 WMMAs / K-step,
// 2 A-frags + 4 B-frags => ~768 B loaded per WMMA).
// Block = 8 waves (2x4) covering 64x256; grid = (N/256, M/64).
// ---------------------------------------------------------------------------
__global__ void __launch_bounds__(256)
wmma_gemm_kernel(const _Float16* __restrict__ A, int lda,
                 const _Float16* __restrict__ W, int ldw,
                 const float* __restrict__ bias,
                 float* __restrict__ outF, int ldf,
                 _Float16* __restrict__ outH, int ldh,
                 int K, int relu) {
  int lane = threadIdx.x & 31;
  int wave = threadIdx.x >> 5;
  int wm = wave >> 2, wn = wave & 3;
  long row0 = (long)blockIdx.y * 64 + wm * 32;   // wave: rows row0..row0+31
  long col0 = (long)blockIdx.x * 256 + wn * 64;  // wave: cols col0..col0+63
  const _Float16* at0 = A + row0 * lda;
  const _Float16* at1 = A + (row0 + 16) * lda;
  const _Float16* wt0 = W + (col0 +  0) * ldw;
  const _Float16* wt1 = W + (col0 + 16) * ldw;
  const _Float16* wt2 = W + (col0 + 32) * ldw;
  const _Float16* wt3 = W + (col0 + 48) * ldw;
  v8f acc[2][4] = {};
  for (int kb = 0; kb < K; kb += 32) {
    v16h a0 = load_frag_a(at0, lda, lane, kb);
    v16h a1 = load_frag_a(at1, lda, lane, kb);
    v16h b0 = load_frag_b(wt0, ldw, lane, kb);
    v16h b1 = load_frag_b(wt1, ldw, lane, kb);
    v16h b2 = load_frag_b(wt2, ldw, lane, kb);
    v16h b3 = load_frag_b(wt3, ldw, lane, kb);
    acc[0][0] = wmma16(a0, b0, acc[0][0]);
    acc[0][1] = wmma16(a0, b1, acc[0][1]);
    acc[0][2] = wmma16(a0, b2, acc[0][2]);
    acc[0][3] = wmma16(a0, b3, acc[0][3]);
    acc[1][0] = wmma16(a1, b0, acc[1][0]);
    acc[1][1] = wmma16(a1, b1, acc[1][1]);
    acc[1][2] = wmma16(a1, b2, acc[1][2]);
    acc[1][3] = wmma16(a1, b3, acc[1][3]);
  }
  int cc = lane & 15;
  int rb = (lane & 16) ? 8 : 0;
#pragma unroll
  for (int ni = 0; ni < 4; ++ni) {
    long col = col0 + ni * 16 + cc;
    float bv = bias ? bias[col] : 0.0f;
#pragma unroll
    for (int mi = 0; mi < 2; ++mi) {
#pragma unroll
      for (int r = 0; r < 8; ++r) {
        float v = acc[mi][ni][r] + bv;
        if (relu) v = fmaxf(v, 0.0f);
        long row = row0 + mi * 16 + rb + r;
        if (outF) outF[row * (long)ldf + col] = v;
        if (outH) outH[row * (long)ldh + col] = (_Float16)v;
      }
    }
  }
}

// ---------------------------------------------------------------------------
// Fused LSTM step: g = xg[:,t,:] + h_prev*Whh^T, gate nonlinearity, c/h update.
// grid = 32 blocks (16 h-cols each), 8 waves = {4 gates} x {2 row tiles}.
// ---------------------------------------------------------------------------
__global__ void __launch_bounds__(256)
lstm_step_kernel(const float* __restrict__ xg,       // [B*T, 4H], row b*T+t
                 const _Float16* __restrict__ Whh,   // [4H, H] f16
                 const _Float16* __restrict__ hprev, // [B, H] f16
                 float* __restrict__ cbuf,           // [B, H] f32
                 _Float16* __restrict__ hnext,       // [B, H] f16 (next step)
                 _Float16* __restrict__ hout, int ldo, // row (b*T+t)*ldo
                 int t) {
  __shared__ float gbuf[4][Bc][16];
  int lane = threadIdx.x & 31;
  int wave = threadIdx.x >> 5;
  int gate = wave >> 1;
  int m0   = (wave & 1) * 16;
  int jb   = blockIdx.x * 16;
  const _Float16* at = hprev + (long)m0 * Hc;
  const _Float16* wt = Whh + (long)(gate * Hc + jb) * Hc;
  v8f acc = {};
  for (int kb = 0; kb < Hc; kb += 32) {
    v16h a = load_frag_a(at, Hc, lane, kb);
    v16h b = load_frag_b(wt, Hc, lane, kb);
    acc = wmma16(a, b, acc);
  }
  int cc = lane & 15;
  int rb = m0 + ((lane & 16) ? 8 : 0);
#pragma unroll
  for (int r = 0; r < 8; ++r) gbuf[gate][rb + r][cc] = acc[r];
  __syncthreads();
  for (int e = threadIdx.x; e < Bc * 16; e += 256) {
    int brow = e >> 4;
    int jj = e & 15;
    int j = jb + jj;
    long xr = ((long)brow * Tc + t) * (4 * Hc);
    float gi = gbuf[0][brow][jj] + xg[xr + j];
    float gf = gbuf[1][brow][jj] + xg[xr + Hc + j];
    float gg = gbuf[2][brow][jj] + xg[xr + 2 * Hc + j];
    float go = gbuf[3][brow][jj] + xg[xr + 3 * Hc + j];
    float cp = cbuf[brow * Hc + j];
    float cn = sigf(gf) * cp + sigf(gi) * tanhf(gg);
    float h  = sigf(go) * tanhf(cn);
    cbuf[brow * Hc + j] = cn;
    hnext[brow * Hc + j] = (_Float16)h;
    hout[((long)brow * Tc + t) * ldo + j] = (_Float16)h;
  }
}

// ---------------------------------------------------------------------------
// Attention scores: scores[b,n,t,s] = q . k / sqrt(HD). grid (S/64, 4, B*NH).
// ---------------------------------------------------------------------------
__global__ void __launch_bounds__(256)
attn_scores_kernel(const _Float16* __restrict__ q,  // [B*T, H], head n at n*HD
                   const _Float16* __restrict__ k,  // [B*S, H]
                   float* __restrict__ scores) {    // [B*NH, T, S]
  int lane = threadIdx.x & 31;
  int wave = threadIdx.x >> 5;
  int wm = wave >> 2, wn = wave & 3;
  int bz = blockIdx.z, b = bz >> 2, n = bz & 3;
  int t0 = blockIdx.y * 32 + wm * 16;
  int s0 = blockIdx.x * 64 + wn * 16;
  int tr = t0 + (lane & 15);
  bool tv = tr < Tc;
  const _Float16* qt = q + ((long)b * Tc + (tv ? tr : 0)) * Hc + n * HDc;
  const _Float16* kt = k + ((long)(b * Sc + s0)) * Hc + n * HDc;
  v8f acc = {};
  for (int kb = 0; kb < HDc; kb += 32) {
    v16h a = {};
    if (tv) {
      int klo = kb + ((lane & 16) ? 8 : 0);
      v8h lo = *(const v8h*)(qt + klo);
      v8h hi = *(const v8h*)(qt + klo + 16);
#pragma unroll
      for (int i = 0; i < 8; ++i) { a[i] = lo[i]; a[i + 8] = hi[i]; }
    }
    v16h bf = load_frag_b(kt, Hc, lane, kb);
    acc = wmma16(a, bf, acc);
  }
  const float scale = 0.08838834764831845f;  // 1/sqrt(128)
  int cc = lane & 15;
  int rb = (lane & 16) ? 8 : 0;
#pragma unroll
  for (int r = 0; r < 8; ++r) {
    int trow = t0 + rb + r;
    if (trow < Tc)
      scores[((long)bz * Tc + trow) * Sc + s0 + cc] = acc[r] * scale;
  }
}

// softmax over S=512 per (b,n,t) row; emits f16 probabilities
__global__ void __launch_bounds__(128)
softmax_kernel(const float* __restrict__ scores, _Float16* __restrict__ p) {
  __shared__ float red[128];
  long row = blockIdx.x;
  const float* src = scores + row * Sc;
  int tid = threadIdx.x;
  float v[4];
  float m = -1e30f;
#pragma unroll
  for (int i = 0; i < 4; ++i) { v[i] = src[tid + i * 128]; m = fmaxf(m, v[i]); }
  red[tid] = m; __syncthreads();
  for (int s = 64; s > 0; s >>= 1) {
    if (tid < s) red[tid] = fmaxf(red[tid], red[tid + s]);
    __syncthreads();
  }
  m = red[0]; __syncthreads();
  float sum = 0.0f;
#pragma unroll
  for (int i = 0; i < 4; ++i) { v[i] = __expf(v[i] - m); sum += v[i]; }
  red[tid] = sum; __syncthreads();
  for (int s = 64; s > 0; s >>= 1) {
    if (tid < s) red[tid] += red[tid + s];
    __syncthreads();
  }
  float inv = 1.0f / red[0];
#pragma unroll
  for (int i = 0; i < 4; ++i) p[row * Sc + tid + i * 128] = (_Float16)(v[i] * inv);
}

// ctx[b,t,n,:] = attn[b,n,t,:] @ v[b,:,n,:]. grid (HD/64, 4, B*NH).
__global__ void __launch_bounds__(256)
attn_ctx_kernel(const _Float16* __restrict__ attnP, // [B*NH, T, S] f16
                const _Float16* __restrict__ vmat,  // [B*S, H]
                _Float16* __restrict__ ctx) {       // [B*T, H]
  int lane = threadIdx.x & 31;
  int wave = threadIdx.x >> 5;
  int wm = wave >> 2, wn = wave & 3;
  int bz = blockIdx.z, b = bz >> 2, n = bz & 3;
  int t0 = blockIdx.y * 32 + wm * 16;
  int j0 = blockIdx.x * 64 + wn * 16;
  int tr = t0 + (lane & 15);
  bool tv = tr < Tc;
  const _Float16* at = attnP + ((long)bz * Tc + (tv ? tr : 0)) * Sc;
  v8f acc = {};
  int jn = j0 + (lane & 15);
  for (int kb = 0; kb < Sc; kb += 32) {
    v16h a = {};
    if (tv) {
      int klo = kb + ((lane & 16) ? 8 : 0);
      v8h lo = *(const v8h*)(at + klo);
      v8h hi = *(const v8h*)(at + klo + 16);
#pragma unroll
      for (int i = 0; i < 8; ++i) { a[i] = lo[i]; a[i + 8] = hi[i]; }
    }
    v16h bf;
    int kg = kb + ((lane & 16) ? 8 : 0);
#pragma unroll
    for (int e = 0; e < 16; ++e)
      bf[e] = vmat[((long)(b * Sc + kg + e + (e & 8))) * Hc + n * HDc + jn];
    acc = wmma16(a, bf, acc);
  }
  int cc = lane & 15;
  int rb = (lane & 16) ? 8 : 0;
#pragma unroll
  for (int r = 0; r < 8; ++r) {
    int trow = t0 + rb + r;
    if (trow < Tc)
      ctx[((long)(b * Tc + trow)) * Hc + n * HDc + j0 + cc] = (_Float16)acc[r];
  }
}

// ---------------------------------------------------------------------------
// small utility kernels
// ---------------------------------------------------------------------------
__global__ void f32_to_f16_kernel(const float* __restrict__ s,
                                  _Float16* __restrict__ d, long n) {
  long i = (long)blockIdx.x * blockDim.x + threadIdx.x;
  if (i < n) d[i] = (_Float16)s[i];
}

__global__ void embed_kernel(const float* __restrict__ emb,
                             const int* __restrict__ tgt,
                             _Float16* __restrict__ x) {
  long row = blockIdx.x;  // 0..B*T-1
  long token = tgt[row];
  const float* e = emb + token * Hc;
  for (int j = threadIdx.x; j < Hc; j += blockDim.x)
    x[row * Hc + j] = (_Float16)e[j];
}

__global__ void bias_sum_kernel(const float* __restrict__ a,
                                const float* __restrict__ b,
                                float* __restrict__ o, int n) {
  int i = blockIdx.x * blockDim.x + threadIdx.x;
  if (i < n) o[i] = a[i] + b[i];
}

__global__ void zero_f32_kernel(float* p, long n) {
  long i = (long)blockIdx.x * blockDim.x + threadIdx.x;
  if (i < n) p[i] = 0.0f;
}
__global__ void zero_f16_kernel(_Float16* p, long n) {
  long i = (long)blockIdx.x * blockDim.x + threadIdx.x;
  if (i < n) p[i] = (_Float16)0.0f;
}

// ---------------------------------------------------------------------------
extern "C" void kernel_launch(void* const* d_in, const int* in_sizes, int n_in,
                              void* d_out, int out_size, void* d_ws, size_t ws_size,
                              hipStream_t stream) {
  const float* enc_in   = (const float*)d_in[0];
  const int*   targets  = (const int*)d_in[1];
  const float* emb      = (const float*)d_in[2];
  const float* Wih0     = (const float*)d_in[3];
  const float* Whh0     = (const float*)d_in[4];
  const float* bih0     = (const float*)d_in[5];
  const float* bhh0     = (const float*)d_in[6];
  const float* Wih1     = (const float*)d_in[7];
  const float* Whh1     = (const float*)d_in[8];
  const float* bih1     = (const float*)d_in[9];
  const float* bhh1     = (const float*)d_in[10];
  const float* attn_in_w  = (const float*)d_in[11];
  const float* attn_in_b  = (const float*)d_in[12];
  const float* attn_out_w = (const float*)d_in[13];
  const float* attn_out_b = (const float*)d_in[14];
  const float* encproj_w  = (const float*)d_in[15];
  const float* encproj_b  = (const float*)d_in[16];
  const float* out1_w     = (const float*)d_in[17];
  const float* out1_b     = (const float*)d_in[18];
  const float* out2_w     = (const float*)d_in[19];
  const float* out2_b     = (const float*)d_in[20];
  float* logits = (float*)d_out;

  // ---- workspace carve-up ----
  char* p = (char*)d_ws;
  auto take = [&](size_t bytes) -> char* {
    char* r = p;
    p += (bytes + 255) & ~(size_t)255;
    return r;
  };
  const long BT = (long)Bc * Tc, BS = (long)Bc * Sc;
  _Float16* enc_in_h = (_Float16*)take((long)Bc * Sc * ENCc * 2);
  _Float16* wpj_h   = (_Float16*)take((long)Hc * ENCc * 2);
  _Float16* wih0_h  = (_Float16*)take((long)4 * Hc * Hc * 2);
  _Float16* whh0_h  = (_Float16*)take((long)4 * Hc * Hc * 2);
  _Float16* wih1_h  = (_Float16*)take((long)4 * Hc * Hc * 2);
  _Float16* whh1_h  = (_Float16*)take((long)4 * Hc * Hc * 2);
  _Float16* wattn_h = (_Float16*)take((long)3 * Hc * Hc * 2);
  _Float16* wao_h   = (_Float16*)take((long)Hc * Hc * 2);
  _Float16* w1_h    = (_Float16*)take((long)Hc * 2 * Hc * 2);
  _Float16* w2_h    = (_Float16*)take((long)Vc * Hc * 2);
  _Float16* x0_h    = (_Float16*)take(BT * Hc * 2);
  _Float16* x1_h    = (_Float16*)take(BT * Hc * 2);
  _Float16* comb_h  = (_Float16*)take(BT * 2 * Hc * 2);
  _Float16* enc_h   = (_Float16*)take(BS * Hc * 2);
  _Float16* q_h     = (_Float16*)take(BT * Hc * 2);
  _Float16* k_h     = (_Float16*)take(BS * Hc * 2);
  _Float16* v_h     = (_Float16*)take(BS * Hc * 2);
  float*    xg      = (float*)take(BT * 4 * Hc * 4);
  float*    scores  = (float*)take((long)Bc * NHc * Tc * Sc * 4);
  _Float16* attnP   = (_Float16*)take((long)Bc * NHc * Tc * Sc * 2);
  _Float16* ctx_h   = (_Float16*)take(BT * Hc * 2);
  _Float16* h1_h    = (_Float16*)take(BT * Hc * 2);
  _Float16* hbuf    = (_Float16*)take((long)2 * Bc * Hc * 2);
  float*    cbuf    = (float*)take((long)Bc * Hc * 4);
  float*    bsum0   = (float*)take(4 * Hc * 4);
  float*    bsum1   = (float*)take(4 * Hc * 4);

  auto conv = [&](const float* s, _Float16* d, long n) {
    f32_to_f16_kernel<<<dim3((unsigned)((n + 255) / 256)), 256, 0, stream>>>(s, d, n);
  };

  // ---- f16 conversions ----
  conv(enc_in, enc_in_h, (long)Bc * Sc * ENCc);
  conv(encproj_w, wpj_h, (long)Hc * ENCc);
  conv(Wih0, wih0_h, (long)4 * Hc * Hc);
  conv(Whh0, whh0_h, (long)4 * Hc * Hc);
  conv(Wih1, wih1_h, (long)4 * Hc * Hc);
  conv(Whh1, whh1_h, (long)4 * Hc * Hc);
  conv(attn_in_w, wattn_h, (long)3 * Hc * Hc);
  conv(attn_out_w, wao_h, (long)Hc * Hc);
  conv(out1_w, w1_h, (long)Hc * 2 * Hc);
  conv(out2_w, w2_h, (long)Vc * Hc);

  embed_kernel<<<dim3((unsigned)BT), 256, 0, stream>>>(emb, targets, x0_h);
  bias_sum_kernel<<<8, 256, 0, stream>>>(bih0, bhh0, bsum0, 4 * Hc);
  bias_sum_kernel<<<8, 256, 0, stream>>>(bih1, bhh1, bsum1, 4 * Hc);

  // ---- encoder projection: [B*S, ENC] x [H, ENC]^T -> enc_h ----
  wmma_gemm_kernel<<<dim3(Hc / 256, (unsigned)(BS / 64)), 256, 0, stream>>>(
      enc_in_h, ENCc, wpj_h, ENCc, encproj_b, nullptr, 0, enc_h, Hc, ENCc, 0);

  // ---- layer 0: xg = x0 * Wih0^T + (bih0+bhh0) ----
  wmma_gemm_kernel<<<dim3(4 * Hc / 256, (unsigned)(BT / 64)), 256, 0, stream>>>(
      x0_h, Hc, wih0_h, Hc, bsum0, xg, 4 * Hc, nullptr, 0, Hc, 0);
  zero_f16_kernel<<<dim3(128), 256, 0, stream>>>(hbuf, (long)2 * Bc * Hc);
  zero_f32_kernel<<<dim3(64), 256, 0, stream>>>(cbuf, (long)Bc * Hc);
  for (int t = 0; t < Tc; ++t)
    lstm_step_kernel<<<dim3(Hc / 16), 256, 0, stream>>>(
        xg, whh0_h, hbuf + (t & 1) * Bc * Hc, cbuf,
        hbuf + ((t + 1) & 1) * Bc * Hc, x1_h, Hc, t);

  // ---- layer 1 (h goes into first half of combined buffer, ld = 2H) ----
  wmma_gemm_kernel<<<dim3(4 * Hc / 256, (unsigned)(BT / 64)), 256, 0, stream>>>(
      x1_h, Hc, wih1_h, Hc, bsum1, xg, 4 * Hc, nullptr, 0, Hc, 0);
  zero_f16_kernel<<<dim3(128), 256, 0, stream>>>(hbuf, (long)2 * Bc * Hc);
  zero_f32_kernel<<<dim3(64), 256, 0, stream>>>(cbuf, (long)Bc * Hc);
  for (int t = 0; t < Tc; ++t)
    lstm_step_kernel<<<dim3(Hc / 16), 256, 0, stream>>>(
        xg, whh1_h, hbuf + (t & 1) * Bc * Hc, cbuf,
        hbuf + ((t + 1) & 1) * Bc * Hc, comb_h, 2 * Hc, t);

  // ---- attention projections ----
  wmma_gemm_kernel<<<dim3(Hc / 256, (unsigned)(BT / 64)), 256, 0, stream>>>(
      comb_h, 2 * Hc, wattn_h, Hc, attn_in_b, nullptr, 0, q_h, Hc, Hc, 0);
  wmma_gemm_kernel<<<dim3(Hc / 256, (unsigned)(BS / 64)), 256, 0, stream>>>(
      enc_h, Hc, wattn_h + (long)Hc * Hc, Hc, attn_in_b + Hc,
      nullptr, 0, k_h, Hc, Hc, 0);
  wmma_gemm_kernel<<<dim3(Hc / 256, (unsigned)(BS / 64)), 256, 0, stream>>>(
      enc_h, Hc, wattn_h + (long)2 * Hc * Hc, Hc, attn_in_b + 2 * Hc,
      nullptr, 0, v_h, Hc, Hc, 0);

  // ---- scores -> softmax -> ctx ----
  attn_scores_kernel<<<dim3(Sc / 64, 4, Bc * NHc), 256, 0, stream>>>(q_h, k_h, scores);
  softmax_kernel<<<dim3(Bc * NHc * Tc), 128, 0, stream>>>(scores, attnP);
  attn_ctx_kernel<<<dim3(HDc / 64, 4, Bc * NHc), 256, 0, stream>>>(attnP, v_h, ctx_h);

  // ---- attn_out -> second half of combined ----
  wmma_gemm_kernel<<<dim3(Hc / 256, (unsigned)(BT / 64)), 256, 0, stream>>>(
      ctx_h, Hc, wao_h, Hc, attn_out_b, nullptr, 0, comb_h + Hc, 2 * Hc, Hc, 0);

  // ---- output head ----
  wmma_gemm_kernel<<<dim3(Hc / 256, (unsigned)(BT / 64)), 256, 0, stream>>>(
      comb_h, 2 * Hc, w1_h, 2 * Hc, out1_b, nullptr, 0, h1_h, Hc, 2 * Hc, 1);
  wmma_gemm_kernel<<<dim3(Vc / 256, (unsigned)(BT / 64)), 256, 0, stream>>>(
      h1_h, Hc, w2_h, Hc, out2_b, logits, Vc, nullptr, 0, Hc, 0);
}
